// Cross_modal_TripletLoss3_36618891166019
// MI455X (gfx1250) — matrix operand order repaired and checked by reading, not compiled, not run on hardware
//
#include <hip/hip_runtime.h>

typedef __attribute__((ext_vector_type(16))) __bf16 v16bf;
typedef __attribute__((ext_vector_type(8)))  __bf16 v8bf;
typedef __attribute__((ext_vector_type(8)))  float  v8f;

#define N_ROWS   4096
#define DIM      128
#define MARGIN_F 0.3f
#define EPS_F    1e-12f

// ---------- bf16 helpers (explicit RNE, no reliance on __bf16 arithmetic) ----------
static __device__ __forceinline__ __bf16 f32_to_bf16(float f) {
  unsigned u = __float_as_uint(f);
  unsigned r = u + 0x7FFFu + ((u >> 16) & 1u);   // round-to-nearest-even
  unsigned short h = (unsigned short)(r >> 16);
  return __builtin_bit_cast(__bf16, h);
}
static __device__ __forceinline__ float bf16_to_f32(__bf16 h) {
  unsigned short s = __builtin_bit_cast(unsigned short, h);
  return __uint_as_float(((unsigned)s) << 16);
}

// ---------- preprocessing: hi/lo bf16 split + row squared norms ----------
__global__ __launch_bounds__(128) void prep_kernel(const float* __restrict__ X,
                                                   __bf16* __restrict__ Xhi,
                                                   __bf16* __restrict__ Xlo,
                                                   float* __restrict__ sq) {
  __shared__ float red[128];
  const int row = blockIdx.x;
  const int t   = threadIdx.x;
  float x = X[row * DIM + t];
  __bf16 h = f32_to_bf16(x);
  float  hf = bf16_to_f32(h);
  __bf16 l = f32_to_bf16(x - hf);
  Xhi[row * DIM + t] = h;
  Xlo[row * DIM + t] = l;
  red[t] = x * x;
  __syncthreads();
  for (int s = 64; s > 0; s >>= 1) {
    if (t < s) red[t] += red[t + s];
    __syncthreads();
  }
  if (t == 0) sq[row] = red[0];
}

__global__ void init_out(float* out) {
  if (threadIdx.x < 2) out[threadIdx.x] = 0.0f;
}

// ---------- WMMA wrappers & operand loaders (layouts per CDNA5 ISA 7.12.2) ----------
static __device__ __forceinline__ v8f wmma_bf16(v16bf a, v16bf b, v8f c) {
  return __builtin_amdgcn_wmma_f32_16x16x32_bf16(false, a, false, b, (short)0, c, false, false);
}

// A operand (16x32 MxK): lane holds row M = lane&15.
// elements e0..7 -> K = kbase + 8*half + e ; e8..15 -> K = kbase + 16 + 8*half + (e-8)
static __device__ __forceinline__ v16bf load_a_op(const __bf16* rowp, int chunk, int half) {
  v8bf x = *(const v8bf*)(rowp + chunk * 32 + 8 * half);
  v8bf y = *(const v8bf*)(rowp + chunk * 32 + 16 + 8 * half);
  return __builtin_shufflevector(x, y, 0, 1, 2, 3, 4, 5, 6, 7, 8, 9, 10, 11, 12, 13, 14, 15);
}

// B operand (32x16 KxN): lane holds column N = lane&15, K = kbase + 8*half + e ; e8..15 -> K = kbase + 16 + 8*half + (e-8).
static __device__ __forceinline__ v16bf load_b_op(const __bf16* rowp, int chunk, int half) {
  v8bf x = *(const v8bf*)(rowp + chunk * 32 + 8 * half);
  v8bf y = *(const v8bf*)(rowp + chunk * 32 + 16 + 8 * half);
  return __builtin_shufflevector(x, y, 0, 1, 2, 3, 4, 5, 6, 7, 8, 9, 10, 11, 12, 13, 14, 15);
}

// ---------- fused distance + hardest-pos/neg + loss kernel ----------
// 8 waves per block, 16 rows per wave -> 128 rows/block, 32 blocks.
__global__ __launch_bounds__(256) void triplet_kernel(
    const __bf16* __restrict__ Ahi, const __bf16* __restrict__ Alo,
    const __bf16* __restrict__ Bhi, const __bf16* __restrict__ Blo,
    const float* __restrict__ sqA, const float* __restrict__ sqB,
    const int* __restrict__ tgt, float* __restrict__ out) {
  const int lane = threadIdx.x & 31;
  const int wave = threadIdx.x >> 5;
  const int half = lane >> 4;
  const int nsub = lane & 15;
  const int rowBase = blockIdx.x * 128 + wave * 16;

  // Hoist A-operand tiles (reused across all 256 column tiles).
  const int aRow = rowBase + nsub;  // A-operand: M = lane&15 for both lane halves
  const __bf16* pAhi = Ahi + aRow * DIM;
  const __bf16* pAlo = Alo + aRow * DIM;
  const __bf16* pBhi = Bhi + aRow * DIM;
  const __bf16* pBlo = Blo + aRow * DIM;
  v16bf aAhi[4], aAlo[4], aBhi[4], aBlo[4];
#pragma unroll
  for (int c = 0; c < 4; ++c) {
    aAhi[c] = load_a_op(pAhi, c, half);
    aAlo[c] = load_a_op(pAlo, c, half);
    aBhi[c] = load_a_op(pBhi, c, half);
    aBlo[c] = load_a_op(pBlo, c, half);
  }

  // Row-side constants per C/D layout: M = rowBase + r + 8*half, N = nsub.
  float sqAm[8], sqBm[8];
  int   tm[8];
#pragma unroll
  for (int r = 0; r < 8; ++r) {
    int m = rowBase + r + 8 * half;
    sqAm[r] = sqA[m];
    sqBm[r] = sqB[m];
    tm[r]   = tgt[m];
  }

  float ap1[8], ap2[8], ap3[8], ap4[8], an1[8], an2[8], an3[8], an4[8];
#pragma unroll
  for (int r = 0; r < 8; ++r) {
    ap1[r] = ap2[r] = ap3[r] = ap4[r] = -3.4e38f;
    an1[r] = an2[r] = an3[r] = an4[r] = 3.4e38f;
  }

  for (int jt = 0; jt < N_ROWS / 16; ++jt) {
    const int n = jt * 16 + nsub;
    const __bf16* qAhi = Ahi + n * DIM;
    const __bf16* qAlo = Alo + n * DIM;
    const __bf16* qBhi = Bhi + n * DIM;
    const __bf16* qBlo = Blo + n * DIM;

    v8f accR = {}, accT = {}, accRT = {}, accTR = {};
#pragma unroll
    for (int c = 0; c < 4; ++c) {
      v16bf bAhi = load_b_op(qAhi, c, half);
      v16bf bAlo = load_b_op(qAlo, c, half);
      v16bf bBhi = load_b_op(qBhi, c, half);
      v16bf bBlo = load_b_op(qBlo, c, half);
      // bf16x3 emulated-f32 products: hi*hi + hi*lo + lo*hi
      accR  = wmma_bf16(aAhi[c], bAhi, accR);
      accR  = wmma_bf16(aAhi[c], bAlo, accR);
      accR  = wmma_bf16(aAlo[c], bAhi, accR);
      accT  = wmma_bf16(aBhi[c], bBhi, accT);
      accT  = wmma_bf16(aBhi[c], bBlo, accT);
      accT  = wmma_bf16(aBlo[c], bBhi, accT);
      accRT = wmma_bf16(aAhi[c], bBhi, accRT);
      accRT = wmma_bf16(aAhi[c], bBlo, accRT);
      accRT = wmma_bf16(aAlo[c], bBhi, accRT);
      accTR = wmma_bf16(aBhi[c], bAhi, accTR);
      accTR = wmma_bf16(aBhi[c], bAlo, accTR);
      accTR = wmma_bf16(aBlo[c], bAhi, accTR);
    }

    const float sqan = sqA[n];
    const float sqbn = sqB[n];
    const int   tn   = tgt[n];
#pragma unroll
    for (int r = 0; r < 8; ++r) {
      const bool same = (tm[r] == tn);
      float dR  = sqAm[r] + sqan - 2.0f * accR[r];
      float dT  = sqBm[r] + sqbn - 2.0f * accT[r];
      float dRT = sqAm[r] + sqbn - 2.0f * accRT[r];
      float dTR = sqBm[r] + sqan - 2.0f * accTR[r];
      if (same) {
        ap1[r] = fmaxf(ap1[r], dR);
        ap2[r] = fmaxf(ap2[r], dT);
        ap3[r] = fmaxf(ap3[r], dRT);
        ap4[r] = fmaxf(ap4[r], dTR);
      } else {
        an1[r] = fminf(an1[r], dR);
        an2[r] = fminf(an2[r], dT);
        an3[r] = fminf(an3[r], dRT);
        an4[r] = fminf(an4[r], dTR);
      }
    }
  }

  // Reduce across the 16 lanes (N dimension) sharing each row; xor masks <16 stay in-half.
#pragma unroll
  for (int r = 0; r < 8; ++r) {
#pragma unroll
    for (int off = 8; off >= 1; off >>= 1) {
      ap1[r] = fmaxf(ap1[r], __shfl_xor(ap1[r], off));
      ap2[r] = fmaxf(ap2[r], __shfl_xor(ap2[r], off));
      ap3[r] = fmaxf(ap3[r], __shfl_xor(ap3[r], off));
      ap4[r] = fmaxf(ap4[r], __shfl_xor(ap4[r], off));
      an1[r] = fminf(an1[r], __shfl_xor(an1[r], off));
      an2[r] = fminf(an2[r], __shfl_xor(an2[r], off));
      an3[r] = fminf(an3[r], __shfl_xor(an3[r], off));
      an4[r] = fminf(an4[r], __shfl_xor(an4[r], off));
    }
  }

  if (nsub == 0) {  // lanes 0 and 16 each own 8 rows
    float loss = 0.0f, prec = 0.0f;
#pragma unroll
    for (int r = 0; r < 8; ++r) {
      float a1 = sqrtf(fmaxf(ap1[r], EPS_F)), n1 = sqrtf(fmaxf(an1[r], EPS_F));
      float a2 = sqrtf(fmaxf(ap2[r], EPS_F)), n2 = sqrtf(fmaxf(an2[r], EPS_F));
      float a3 = sqrtf(fmaxf(ap3[r], EPS_F)), n3 = sqrtf(fmaxf(an3[r], EPS_F));
      float a4 = sqrtf(fmaxf(ap4[r], EPS_F)), n4 = sqrtf(fmaxf(an4[r], EPS_F));
      loss += fmaxf(a1 - n1 + MARGIN_F, 0.0f) + fmaxf(a2 - n2 + MARGIN_F, 0.0f) +
              fmaxf(a3 - n3 + MARGIN_F, 0.0f) + fmaxf(a4 - n4 + MARGIN_F, 0.0f) +
              fmaxf(a3 - n1 + MARGIN_F, 0.0f) + fmaxf(a4 - n2 + MARGIN_F, 0.0f);
      prec += (float)(n1 > a1) + (float)(n2 > a2) + (float)(n3 > a3) +
              (float)(n4 > a4) + (float)(n1 > a3) + (float)(n2 > a4);
    }
    const float inv = 1.0f / (6.0f * (float)N_ROWS);
    atomicAdd(&out[0], loss * inv);
    atomicAdd(&out[1], prec * inv);
  }
}

// ---------- host launch ----------
extern "C" void kernel_launch(void* const* d_in, const int* in_sizes, int n_in,
                              void* d_out, int out_size, void* d_ws, size_t ws_size,
                              hipStream_t stream) {
  const float* m1  = (const float*)d_in[0];  // [4096,128] f32
  const float* m2  = (const float*)d_in[1];  // [4096,128] f32
  const int*   tgt = (const int*)d_in[2];    // [4096] int
  float*       out = (float*)d_out;          // [loss, prec]

  // Workspace layout (needs ~4.2 MB): 4 bf16 matrices + 2 norm vectors.
  const size_t mat_elems = (size_t)N_ROWS * DIM;
  __bf16* Ahi = (__bf16*)d_ws;
  __bf16* Alo = Ahi + mat_elems;
  __bf16* Bhi = Alo + mat_elems;
  __bf16* Blo = Bhi + mat_elems;
  float*  sqA = (float*)(Blo + mat_elems);
  float*  sqB = sqA + N_ROWS;
  (void)ws_size; (void)in_sizes; (void)n_in; (void)out_size;

  prep_kernel<<<N_ROWS, 128, 0, stream>>>(m1, Ahi, Alo, sqA);
  prep_kernel<<<N_ROWS, 128, 0, stream>>>(m2, Bhi, Blo, sqB);
  init_out<<<1, 32, 0, stream>>>(out);
  // 32 blocks * 8 waves * 16 rows = 4096 rows
  triplet_kernel<<<N_ROWS / 128, 256, 0, stream>>>(Ahi, Alo, Bhi, Blo, sqA, sqB, tgt, out);
}
